// GraphQA_88287347736840
// MI455X (gfx1250) — matrix operand order repaired and checked by reading, not compiled, not run on hardware
//
#include <hip/hip_runtime.h>
#include <math.h>

typedef __attribute__((ext_vector_type(16))) _Float16 v16h;
typedef __attribute__((ext_vector_type(8)))  _Float16 v8h;
typedef __attribute__((ext_vector_type(8)))  float    v8f;

#define F_DIM 128
#define N_O   64
#define N_E   512
#define B_SZ  32
#define T_STEPS 16
#define EPSV 1e-5f

union FragAB { v16h v; v8h h[2]; };

// ---------------- one-time prep kernels ----------------

// Recover src/dst indices from one-hot incidence matrices (N_O x N_E).
__global__ void decode_idx_k(const float* __restrict__ Rs, const float* __restrict__ Rr,
                             int* __restrict__ src_idx, int* __restrict__ dst_idx) {
    int e = threadIdx.x;             // 512 threads, 1 block
    int s = 0, d = 0;
    for (int o = 0; o < N_O; ++o) {
        if (Rs[o * N_E + e] > 0.5f) s = o;
        if (Rr[o * N_E + e] > 0.5f) d = o;
    }
    src_idx[e] = s;
    dst_idx[e] = d;
}

// Serial, deterministic CSR build for the destination scatter (stable by e).
__global__ void csr_build_k(const int* __restrict__ dst_idx,
                            int* __restrict__ off, int* __restrict__ lst) {
    if (threadIdx.x != 0 || blockIdx.x != 0) return;
    int cnt[N_O];
    for (int o = 0; o < N_O; ++o) cnt[o] = 0;
    for (int e = 0; e < N_E; ++e) cnt[dst_idx[e]]++;
    off[0] = 0;
    for (int o = 0; o < N_O; ++o) off[o + 1] = off[o] + cnt[o];
    int cur[N_O];
    for (int o = 0; o < N_O; ++o) cur[o] = off[o];
    for (int e = 0; e < N_E; ++e) lst[cur[dst_idx[e]]++] = e;
}

// Convert W1/W2 (2F x F, row-major) to f16, transposed to (F x 2F) so the
// WMMA B-fragment (K contiguous per output column) is two contiguous v8h loads.
__global__ void prep_weights_k(const float* __restrict__ W1, const float* __restrict__ W2,
                               _Float16* __restrict__ Wt1, _Float16* __restrict__ Wt2) {
    int idx = blockIdx.x * blockDim.x + threadIdx.x;   // 65536
    const float* W = (idx < 32768) ? W1 : W2;
    _Float16* Wt   = (idx < 32768) ? Wt1 : Wt2;
    int r = idx & 32767;
    int n = r >> 8;        // output col 0..127
    int c = r & 255;       // input channel 0..255
    Wt[r] = (_Float16)W[c * F_DIM + n];
}

__global__ void copy_f32_k(float* __restrict__ dst, const float* __restrict__ src, int n) {
    for (int i = blockIdx.x * blockDim.x + threadIdx.x; i < n; i += gridDim.x * blockDim.x)
        dst[i] = src[i];
}

// ---------------- per-step kernels ----------------

// Per-node Σx, Σx² over (B, F)  -> BN1 stats building blocks.
__global__ void bn1_stats_k(const float* __restrict__ x,
                            float* __restrict__ S, float* __restrict__ SS) {
    __shared__ float s1[256], s2[256];
    int o = blockIdx.x, tid = threadIdx.x;
    float a = 0.f, b = 0.f;
    for (int i = tid; i < B_SZ * F_DIM; i += 256) {
        int bb = i >> 7, f = i & 127;
        float v = x[(bb * N_O + o) * F_DIM + f];
        a += v; b += v * v;
    }
    s1[tid] = a; s2[tid] = b; __syncthreads();
    for (int st = 128; st > 0; st >>= 1) {
        if (tid < st) { s1[tid] += s1[tid + st]; s2[tid] += s2[tid + st]; }
        __syncthreads();
    }
    if (tid == 0) { S[o] = s1[0]; SS[o] = s2[0]; }
}

// BN1 channel stats: each channel mixes 4 fixed nodes' full (B,F) distributions.
__global__ void bn1_final_k(const float* __restrict__ S, const float* __restrict__ SS,
                            const int* __restrict__ src_idx, const int* __restrict__ dst_idx,
                            const float* __restrict__ g1, const float* __restrict__ be1,
                            float* __restrict__ scale1, float* __restrict__ shift1) {
    int c = threadIdx.x;   // 256 threads, 1 block
    float tot = 0.f, tot2 = 0.f;
    #pragma unroll
    for (int el = 0; el < 4; ++el) {
        int node = (el < 2) ? src_idx[el * 256 + c] : dst_idx[(el - 2) * 256 + c];
        tot += S[node]; tot2 += SS[node];
    }
    const float inv = 1.f / (float)(B_SZ * N_E);
    float mean = tot * inv;
    float var  = tot2 * inv - mean * mean;
    float sc   = g1[c] * rsqrtf(var + EPSV);
    scale1[c] = sc;
    shift1[c] = be1[c] - mean * sc;
}

// GEMM1: edge_obj = relu(BN(node_obj)) @ W1 + b1.  M=16384, K=256, N=128.
// 4 waves/block, 1 M-tile (16 rows) per wave; A staged in LDS as f16.
__global__ void gemm1_k(const float* __restrict__ x,
                        const int* __restrict__ src_idx, const int* __restrict__ dst_idx,
                        const float* __restrict__ scale1, const float* __restrict__ shift1,
                        const _Float16* __restrict__ Wt1, const float* __restrict__ b1,
                        float* __restrict__ edge_obj) {
    __shared__ _Float16 ldsA[4 * 16 * 256];
    int tid = threadIdx.x;
    int w = tid >> 5, lane = tid & 31;
    int mtBase = blockIdx.x * 4;

    // Fill: A[m][c] = relu(x[b, lut[m&3][c], e0/4 + (m>>2)] * scale1[c] + shift1[c])
    for (int idx = tid; idx < 4 * 4096; idx += 128) {
        int tw = idx >> 12;
        int r  = idx & 4095;
        int m  = r >> 8;
        int c  = r & 255;
        int rowbase = (mtBase + tw) << 4;
        int b  = rowbase >> 9;
        int e0 = rowbase & 511;
        int f  = (e0 >> 2) + (m >> 2);
        int el = m & 3;
        int node = (el < 2) ? src_idx[el * 256 + c] : dst_idx[(el - 2) * 256 + c];
        float v = x[(b * N_O + node) * F_DIM + f] * scale1[c] + shift1[c];
        ldsA[idx] = (_Float16)fmaxf(v, 0.f);
    }
    __syncthreads();

    int rowbase = (mtBase + w) << 4;
    int g = lane >> 4, m16 = lane & 15;
    const _Float16* myA = &ldsA[w * 4096];

    FragAB A[8];
    #pragma unroll
    for (int k = 0; k < 8; ++k) {
        int off = m16 * 256 + k * 32 + g * 8;
        A[k].h[0] = *(const v8h*)&myA[off];
        A[k].h[1] = *(const v8h*)&myA[off + 16];
    }

    #pragma unroll
    for (int nt = 0; nt < 8; ++nt) {
        v8f acc;
        #pragma unroll
        for (int r = 0; r < 8; ++r) acc[r] = 0.f;
        #pragma unroll
        for (int k = 0; k < 8; ++k) {
            FragAB Bf;
            const _Float16* bp = Wt1 + (nt * 16 + m16) * 256 + k * 32 + g * 8;
            Bf.h[0] = *(const v8h*)bp;
            Bf.h[1] = *(const v8h*)(bp + 16);
            acc = __builtin_amdgcn_wmma_f32_16x16x32_f16(false, A[k].v, false, Bf.v,
                                                         (short)0, acc, false, false);
        }
        int col = nt * 16 + m16;
        float bias = b1[col];
        #pragma unroll
        for (int r = 0; r < 8; ++r) {
            int row = rowbase + g * 8 + r;
            edge_obj[row * F_DIM + col] = acc[r] + bias;
        }
    }
}

// Deterministic scatter-sum to dst nodes via CSR (edge order fixed).
__global__ void agg_k(const float* __restrict__ edge_obj,
                      const int* __restrict__ off, const int* __restrict__ lst,
                      float* __restrict__ agg) {
    int o = blockIdx.x, b = blockIdx.y, f = threadIdx.x;
    int s = off[o], e = off[o + 1];
    float acc = 0.f;
    for (int i = s; i < e; ++i)
        acc += edge_obj[(b * N_E + lst[i]) * F_DIM + f];
    agg[(b * N_O + o) * F_DIM + f] = acc;
}

// BN2 stats over upd = concat([x, agg]) rows (B*N_O = 2048).
__global__ void bn2_stats_k(const float* __restrict__ cur, const float* __restrict__ agg,
                            const float* __restrict__ g2, const float* __restrict__ be2,
                            float* __restrict__ scale2, float* __restrict__ shift2) {
    __shared__ float s1[128], s2[128];
    int c = blockIdx.x, tid = threadIdx.x;
    const float* src = (c < F_DIM) ? cur : agg;
    int cc = c & 127;
    float a = 0.f, b = 0.f;
    for (int r = tid; r < B_SZ * N_O; r += 128) {
        float v = src[r * F_DIM + cc];
        a += v; b += v * v;
    }
    s1[tid] = a; s2[tid] = b; __syncthreads();
    for (int st = 64; st > 0; st >>= 1) {
        if (tid < st) { s1[tid] += s1[tid + st]; s2[tid] += s2[tid + st]; }
        __syncthreads();
    }
    if (tid == 0) {
        const float inv = 1.f / (float)(B_SZ * N_O);
        float mean = s1[0] * inv;
        float var  = s2[0] * inv - mean * mean;
        float sc   = g2[c] * rsqrtf(var + EPSV);
        scale2[c] = sc;
        shift2[c] = be2[c] - mean * sc;
    }
}

// GEMM2: ent_next = relu(BN(concat([x, agg]))) @ W2 + b2.  M=2048, K=256, N=128.
__global__ void gemm2_k(const float* __restrict__ cur, const float* __restrict__ agg,
                        const float* __restrict__ scale2, const float* __restrict__ shift2,
                        const _Float16* __restrict__ Wt2, const float* __restrict__ b2,
                        float* __restrict__ nxt) {
    __shared__ _Float16 ldsA[4 * 16 * 256];
    int tid = threadIdx.x;
    int w = tid >> 5, lane = tid & 31;
    int mtBase = blockIdx.x * 4;

    for (int idx = tid; idx < 4 * 4096; idx += 128) {
        int tw = idx >> 12;
        int r  = idx & 4095;
        int m  = r >> 8;
        int c  = r & 255;
        int row = ((mtBase + tw) << 4) + m;      // row = b*64 + o
        float v = (c < F_DIM) ? cur[row * F_DIM + c]
                              : agg[row * F_DIM + (c - F_DIM)];
        v = v * scale2[c] + shift2[c];
        ldsA[idx] = (_Float16)fmaxf(v, 0.f);
    }
    __syncthreads();

    int rowbase = (mtBase + w) << 4;
    int g = lane >> 4, m16 = lane & 15;
    const _Float16* myA = &ldsA[w * 4096];

    FragAB A[8];
    #pragma unroll
    for (int k = 0; k < 8; ++k) {
        int off = m16 * 256 + k * 32 + g * 8;
        A[k].h[0] = *(const v8h*)&myA[off];
        A[k].h[1] = *(const v8h*)&myA[off + 16];
    }

    #pragma unroll
    for (int nt = 0; nt < 8; ++nt) {
        v8f acc;
        #pragma unroll
        for (int r = 0; r < 8; ++r) acc[r] = 0.f;
        #pragma unroll
        for (int k = 0; k < 8; ++k) {
            FragAB Bf;
            const _Float16* bp = Wt2 + (nt * 16 + m16) * 256 + k * 32 + g * 8;
            Bf.h[0] = *(const v8h*)bp;
            Bf.h[1] = *(const v8h*)(bp + 16);
            acc = __builtin_amdgcn_wmma_f32_16x16x32_f16(false, A[k].v, false, Bf.v,
                                                         (short)0, acc, false, false);
        }
        int col = nt * 16 + m16;
        float bias = b2[col];
        #pragma unroll
        for (int r = 0; r < 8; ++r) {
            int row = rowbase + g * 8 + r;
            nxt[row * F_DIM + col] = acc[r] + bias;
        }
    }
}

// Location head: (B,6,F)@(F,8)+bloc, sigmoid on last 4 channels.
__global__ void loc_k(const float* __restrict__ ent, const float* __restrict__ Wloc,
                      const float* __restrict__ bloc, float* __restrict__ out, int t) {
    int b = blockIdx.x, tid = threadIdx.x;
    if (tid >= 48) return;
    int i = tid >> 3, j = tid & 7;
    float acc = bloc[j];
    for (int f = 0; f < F_DIM; ++f)
        acc += ent[(b * N_O + i) * F_DIM + f] * Wloc[f * 8 + j];
    if (j >= 4) acc = 1.f / (1.f + expf(-acc));
    out[((b * T_STEPS + t) * 6 + i) * 8 + j] = acc;
}

// ---------------- launch ----------------

extern "C" void kernel_launch(void* const* d_in, const int* in_sizes, int n_in,
                              void* d_out, int out_size, void* d_ws, size_t ws_size,
                              hipStream_t stream) {
    const float* entity = (const float*)d_in[0];
    const float* Rs   = (const float*)d_in[1];
    const float* Rr   = (const float*)d_in[2];
    const float* W1   = (const float*)d_in[3];
    const float* b1   = (const float*)d_in[4];
    const float* g1   = (const float*)d_in[5];
    const float* be1  = (const float*)d_in[6];
    const float* W2   = (const float*)d_in[7];
    const float* b2   = (const float*)d_in[8];
    const float* g2   = (const float*)d_in[9];
    const float* be2  = (const float*)d_in[10];
    const float* Wloc = (const float*)d_in[11];
    const float* bloc = (const float*)d_in[12];

    float* out_loc  = (float*)d_out;                       // (B,T,6,8) = 24576
    float* out_last = (float*)d_out + B_SZ * T_STEPS * 6 * 8;

    char* ws = (char*)d_ws;
    int*  src_idx = (int*)(ws + 0);
    int*  dst_idx = (int*)(ws + 2048);
    int*  csr_off = (int*)(ws + 4096);
    int*  csr_lst = (int*)(ws + 4608);
    float* Sn     = (float*)(ws + 8192);
    float* SSn    = (float*)(ws + 8448);
    float* scale1 = (float*)(ws + 8704);
    float* shift1 = (float*)(ws + 9728);
    float* scale2 = (float*)(ws + 10752);
    float* shift2 = (float*)(ws + 11776);
    _Float16* Wt1 = (_Float16*)(ws + 16384);
    _Float16* Wt2 = (_Float16*)(ws + 81920);
    float* entA   = (float*)(ws + 147456);
    float* entB   = (float*)(ws + 147456 + 4194304 / 4);   // +1MB
    float* agg    = (float*)(ws + 147456 + 2 * 1048576);
    float* edge_obj = (float*)(ws + 147456 + 3 * 1048576);

    const int ENT_N = B_SZ * N_O * F_DIM;                  // 262144

    decode_idx_k<<<1, 512, 0, stream>>>(Rs, Rr, src_idx, dst_idx);
    csr_build_k<<<1, 32, 0, stream>>>(dst_idx, csr_off, csr_lst);
    prep_weights_k<<<256, 256, 0, stream>>>(W1, W2, Wt1, Wt2);
    copy_f32_k<<<256, 256, 0, stream>>>(entA, entity, ENT_N);

    float* bufs[2] = {entA, entB};
    for (int t = 0; t < T_STEPS; ++t) {
        float* cur = bufs[t & 1];
        float* nxt = bufs[(t + 1) & 1];

        bn1_stats_k<<<N_O, 256, 0, stream>>>(cur, Sn, SSn);
        bn1_final_k<<<1, 256, 0, stream>>>(Sn, SSn, src_idx, dst_idx, g1, be1, scale1, shift1);
        gemm1_k<<<256, 128, 0, stream>>>(cur, src_idx, dst_idx, scale1, shift1, Wt1, b1, edge_obj);
        agg_k<<<dim3(N_O, B_SZ), F_DIM, 0, stream>>>(edge_obj, csr_off, csr_lst, agg);
        bn2_stats_k<<<256, 128, 0, stream>>>(cur, agg, g2, be2, scale2, shift2);
        gemm2_k<<<32, 128, 0, stream>>>(cur, agg, scale2, shift2, Wt2, b2, nxt);
        loc_k<<<B_SZ, 64, 0, stream>>>(nxt, Wloc, bloc, out_loc, t);
    }
    copy_f32_k<<<256, 256, 0, stream>>>(out_last, bufs[0], ENT_N);
}
